// MoEST_Plus_Inference_81922206204360
// MI455X (gfx1250) — compile-verified, matrix operand order checked by reading, not run on hardware
//
#include <hip/hip_runtime.h>
#include <hip/hip_bf16.h>
#include <math.h>

#define NTOK  4096
#define DUIN  1024
#define HID   256
#define NEXP  4
#define NGENE 2000
#define NHEAD 4
#define HDIM  64

typedef __attribute__((ext_vector_type(16))) __bf16 v16bf;
typedef __attribute__((ext_vector_type(8)))  float  v8f;

union Frag { v16bf bf; uint4 q[2]; unsigned short us[16]; };

// native bf16 converts (RNE) -> v_cvt_pk_bf16_f32 on gfx1250
__device__ __forceinline__ unsigned int cvt2_bf16(float x, float y) {
  union { __bf16 b[2]; unsigned int u; } r;
  r.b[0] = (__bf16)x; r.b[1] = (__bf16)y;
  return r.u;
}
__device__ __forceinline__ unsigned short cvt1_bf16(float x) {
  union { __bf16 b; unsigned short u; } r;
  r.b = (__bf16)x; return r.u;
}

__device__ __forceinline__ float gelu_exact(float x) {
  return 0.5f * x * (1.0f + erff(x * 0.70710678118654752f));
}
__device__ __forceinline__ float softplus_f(float x) {
  return fmaxf(x, 0.0f) + log1pf(expf(-fabsf(x)));
}

// ---------------------------------------------------------------------------
// Generic bf16-WMMA GEMM: C[M,N] = epi(A[M,K] @ W[K,N])
// block = 128 threads (4 waves); block tile 64x32; K staged in LDS by 32.
// Tiles stored row-major in k; each WMMA operand = 2x ds_load_b128
// (half-group k ranges half*8..+7 and 16+half*8..+7 are naturally contiguous).
// epilogue: +bias, act(0 none,1 gelu,2 softplus), *rowscale[m], +resid, accum
// ---------------------------------------------------------------------------
__global__ __launch_bounds__(128)
void gemm_bf16_kernel(const float* __restrict__ A, const float* __restrict__ W,
                      const float* __restrict__ bias, const float* __restrict__ resid,
                      const float* __restrict__ rowscale, float* __restrict__ C,
                      int M, int N, int K, int act, int accum)
{
  __shared__ uint4 sAq[(64 * 32) / 8];       // 64 rows x 32 k (bf16)
  __shared__ uint4 sWq[(32 * 32) / 8];       // 32 cols x 32 k (bf16, transposed)
  unsigned short* sA = (unsigned short*)sAq;
  unsigned short* sW = (unsigned short*)sWq;

  const int tid  = threadIdx.x;
  const int wave = tid >> 5;
  const int lane = tid & 31;
  const int half = lane >> 4;
  const int lidx = lane & 15;
  const int row0 = blockIdx.x * 64;
  const int col0 = blockIdx.y * 32;

  v8f acc0 = {}, acc1 = {};

  for (int k0 = 0; k0 < K; k0 += 32) {
    // A tile: float4 coalesced loads, packed bf16 8-byte LDS stores
    for (int i = tid; i < 64 * 8; i += 128) {
      int r = i >> 3, k4 = (i & 7) << 2;
      const float4 f = *(const float4*)(A + (size_t)(row0 + r) * K + k0 + k4);
      uint2 wv;
      wv.x = cvt2_bf16(f.x, f.y);
      wv.y = cvt2_bf16(f.z, f.w);
      *(uint2*)(sA + r * 32 + k4) = wv;
    }
    // W tile: float4 coalesced loads along N, transposed into LDS
    for (int i = tid; i < 32 * 8; i += 128) {
      int kk = i >> 3, c4 = (i & 7) << 2;
      int gc = col0 + c4;
      float4 f;
      if (gc < N) f = *(const float4*)(W + (size_t)(k0 + kk) * N + gc);
      else { f.x = 0.f; f.y = 0.f; f.z = 0.f; f.w = 0.f; }
      sW[(c4 + 0) * 32 + kk] = cvt1_bf16(f.x);
      sW[(c4 + 1) * 32 + kk] = cvt1_bf16(f.y);
      sW[(c4 + 2) * 32 + kk] = cvt1_bf16(f.z);
      sW[(c4 + 3) * 32 + kk] = cvt1_bf16(f.w);
    }
    __syncthreads();

    Frag a, b0, b1;
    { const unsigned short* p = sA + (wave * 16 + lidx) * 32 + half * 8;
      a.q[0] = *(const uint4*)p; a.q[1] = *(const uint4*)(p + 16); }
    { const unsigned short* p = sW + lidx * 32 + half * 8;
      b0.q[0] = *(const uint4*)p; b0.q[1] = *(const uint4*)(p + 16); }
    { const unsigned short* p = sW + (16 + lidx) * 32 + half * 8;
      b1.q[0] = *(const uint4*)p; b1.q[1] = *(const uint4*)(p + 16); }

    acc0 = __builtin_amdgcn_wmma_f32_16x16x32_bf16(false, a.bf, false, b0.bf,
                                                   (short)0, acc0, false, false);
    acc1 = __builtin_amdgcn_wmma_f32_16x16x32_bf16(false, a.bf, false, b1.bf,
                                                   (short)0, acc1, false, false);
    __syncthreads();
  }

#pragma unroll
  for (int j = 0; j < 8; ++j) {
    int m = row0 + wave * 16 + j + half * 8;
#pragma unroll
    for (int c = 0; c < 2; ++c) {
      int n = col0 + c * 16 + lidx;
      if (n >= N) continue;
      float x = (c == 0) ? acc0[j] : acc1[j];
      if (bias) x += bias[n];
      if (act == 1)      x = gelu_exact(x);
      else if (act == 2) x = softplus_f(x);
      if (rowscale) x *= rowscale[m];
      if (resid)    x += resid[(size_t)m * N + n];
      size_t off = (size_t)m * N + n;
      if (accum) C[off] += x; else C[off] = x;
    }
  }
}

// ---------------------------------------------------------------------------
// Flash attention: grid(N/64, NHEAD), block 128 (4 waves x 16 Q rows).
// K/V streamed in 32-key chunks via LDS (bf16); online softmax in f32.
// ---------------------------------------------------------------------------
__global__ __launch_bounds__(128)
void attn_kernel(const float* __restrict__ q, const float* __restrict__ k,
                 const float* __restrict__ v, float* __restrict__ ao)
{
  __shared__ uint4 sKq[(32 * 64) / 8];       // [key][d]            (bf16)
  __shared__ uint4 sVq[(32 * 64) / 8];       // [col][key]          (bf16, transposed)
  __shared__ uint4 sPq[4][(16 * 32) / 8];    // per-wave P tile     (bf16, [row][key])
  unsigned short* sK = (unsigned short*)sKq;
  unsigned short* sV = (unsigned short*)sVq;

  const int tid  = threadIdx.x;
  const int wave = tid >> 5;
  const int lane = tid & 31;
  const int half = lane >> 4;
  const int lidx = lane & 15;
  const int h    = blockIdx.y;
  const int q0   = blockIdx.x * 64 + wave * 16;
  unsigned short* sPw = (unsigned short*)sPq[wave];

  Frag qa0, qa1;
  {
    const float* qrow = q + (size_t)(q0 + lidx) * HID + h * HDIM;
#pragma unroll
    for (int e = 0; e < 8; ++e) {
      qa0.us[e]     = cvt1_bf16(qrow[half * 8 + e] * 0.125f);      // 1/sqrt(64)
      qa0.us[8 + e] = cvt1_bf16(qrow[16 + half * 8 + e] * 0.125f);
      qa1.us[e]     = cvt1_bf16(qrow[32 + half * 8 + e] * 0.125f);
      qa1.us[8 + e] = cvt1_bf16(qrow[48 + half * 8 + e] * 0.125f);
    }
  }

  float mrow[8], lrow[8];
#pragma unroll
  for (int j = 0; j < 8; ++j) { mrow[j] = -INFINITY; lrow[j] = 0.0f; }
  v8f o[4] = {};

  for (int kb = 0; kb < NTOK; kb += 32) {
    for (int i = tid; i < 32 * 16; i += 128) {
      int kr = i >> 4, d4 = (i & 15) << 2;
      const float4 fk = *(const float4*)(k + (size_t)(kb + kr) * HID + h * HDIM + d4);
      const float4 fv = *(const float4*)(v + (size_t)(kb + kr) * HID + h * HDIM + d4);
      uint2 wk;
      wk.x = cvt2_bf16(fk.x, fk.y);
      wk.y = cvt2_bf16(fk.z, fk.w);
      *(uint2*)(sK + kr * 64 + d4) = wk;
      sV[(d4 + 0) * 32 + kr] = cvt1_bf16(fv.x);
      sV[(d4 + 1) * 32 + kr] = cvt1_bf16(fv.y);
      sV[(d4 + 2) * 32 + kr] = cvt1_bf16(fv.z);
      sV[(d4 + 3) * 32 + kr] = cvt1_bf16(fv.w);
    }
    __syncthreads();

    // S = Q @ K^T  (two 16-key subtiles x two 32-dim chunks)
    v8f s0 = {}, s1 = {};
    {
      Frag b;
      { const unsigned short* p = sK + lidx * 64 + half * 8;
        b.q[0] = *(const uint4*)p; b.q[1] = *(const uint4*)(p + 16); }
      s0 = __builtin_amdgcn_wmma_f32_16x16x32_bf16(false, qa0.bf, false, b.bf, (short)0, s0, false, false);
      { const unsigned short* p = sK + lidx * 64 + 32 + half * 8;
        b.q[0] = *(const uint4*)p; b.q[1] = *(const uint4*)(p + 16); }
      s0 = __builtin_amdgcn_wmma_f32_16x16x32_bf16(false, qa1.bf, false, b.bf, (short)0, s0, false, false);
      { const unsigned short* p = sK + (16 + lidx) * 64 + half * 8;
        b.q[0] = *(const uint4*)p; b.q[1] = *(const uint4*)(p + 16); }
      s1 = __builtin_amdgcn_wmma_f32_16x16x32_bf16(false, qa0.bf, false, b.bf, (short)0, s1, false, false);
      { const unsigned short* p = sK + (16 + lidx) * 64 + 32 + half * 8;
        b.q[0] = *(const uint4*)p; b.q[1] = *(const uint4*)(p + 16); }
      s1 = __builtin_amdgcn_wmma_f32_16x16x32_bf16(false, qa1.bf, false, b.bf, (short)0, s1, false, false);
    }

    // online softmax (row reductions across the 16-lane half-groups)
#pragma unroll
    for (int j = 0; j < 8; ++j) {
      float mx = fmaxf(s0[j], s1[j]);
#pragma unroll
      for (int off = 1; off < 16; off <<= 1) mx = fmaxf(mx, __shfl_xor(mx, off, 32));
      float nm = fmaxf(mrow[j], mx);
      float sc = expf(mrow[j] - nm);
      float p0 = expf(s0[j] - nm);
      float p1 = expf(s1[j] - nm);
      float rs = p0 + p1;
#pragma unroll
      for (int off = 1; off < 16; off <<= 1) rs += __shfl_xor(rs, off, 32);
      lrow[j] = lrow[j] * sc + rs;
      mrow[j] = nm;
      o[0][j] *= sc; o[1][j] *= sc; o[2][j] *= sc; o[3][j] *= sc;
      // stash P as bf16 directly, [row][key] layout
      sPw[(j + half * 8) * 32 + lidx]      = cvt1_bf16(p0);
      sPw[(j + half * 8) * 32 + 16 + lidx] = cvt1_bf16(p1);
    }
    __syncthreads();

    // O += P @ V
    Frag pa;
    { const unsigned short* p = sPw + lidx * 32 + half * 8;
      pa.q[0] = *(const uint4*)p; pa.q[1] = *(const uint4*)(p + 16); }
#pragma unroll
    for (int c = 0; c < 4; ++c) {
      Frag b;
      { const unsigned short* p = sV + (c * 16 + lidx) * 32 + half * 8;
        b.q[0] = *(const uint4*)p; b.q[1] = *(const uint4*)(p + 16); }
      o[c] = __builtin_amdgcn_wmma_f32_16x16x32_bf16(false, pa.bf, false, b.bf,
                                                     (short)0, o[c], false, false);
    }
    __syncthreads();
  }

#pragma unroll
  for (int j = 0; j < 8; ++j) {
    float inv = 1.0f / lrow[j];
    int m = q0 + j + half * 8;
    float* arow = ao + (size_t)m * HID + h * HDIM;
    arow[lidx]      = o[0][j] * inv;
    arow[16 + lidx] = o[1][j] * inv;
    arow[32 + lidx] = o[2][j] * inv;
    arow[48 + lidx] = o[3][j] * inv;
  }
}

// ---------------------------------------------------------------------------
// small VALU kernels
// ---------------------------------------------------------------------------
__global__ void fourier_kernel(const float* __restrict__ pos, const float* __restrict__ B,
                               float* __restrict__ four)
{
  int i = blockIdx.x * blockDim.x + threadIdx.x;
  if (i >= NTOK * 64) return;
  int n = i >> 6, c = i & 63;
  float x = 6.283185307179586f *
            (pos[n * 3 + 0] * B[c] + pos[n * 3 + 1] * B[64 + c] + pos[n * 3 + 2] * B[128 + c]);
  four[(size_t)n * 128 + c]      = sinf(x);
  four[(size_t)n * 128 + 64 + c] = cosf(x);
}

__global__ __launch_bounds__(256)
void ln_kernel(float* __restrict__ buf, const float* __restrict__ g,
               const float* __restrict__ b, int rows, int dogelu)
{
  int row = blockIdx.x * 8 + (threadIdx.x >> 5);
  if (row >= rows) return;
  int lane = threadIdx.x & 31;
  float* p = buf + (size_t)row * HID;
  float s = 0.f, s2 = 0.f;
  for (int i = lane; i < HID; i += 32) { float x = p[i]; s += x; s2 += x * x; }
#pragma unroll
  for (int off = 16; off; off >>= 1) { s += __shfl_xor(s, off, 32); s2 += __shfl_xor(s2, off, 32); }
  float mean = s * (1.0f / HID);
  float var  = s2 * (1.0f / HID) - mean * mean;
  float inv  = rsqrtf(var + 1e-5f);
  for (int i = lane; i < HID; i += 32) {
    float x = (p[i] - mean) * inv * g[i] + b[i];
    if (dogelu) x = gelu_exact(x);
    p[i] = x;
  }
}

__global__ void router_kernel(const float* __restrict__ z, const float* __restrict__ grad,
                              const float* __restrict__ Wr, const float* __restrict__ br,
                              float* __restrict__ probs_out, float* __restrict__ gate)
{
  int n = blockIdx.x * blockDim.x + threadIdx.x;
  if (n >= NTOK) return;
  float lg[NEXP];
#pragma unroll
  for (int e = 0; e < NEXP; ++e) lg[e] = br[e];
  const float* zr = z + (size_t)n * HID;
  for (int kk = 0; kk < HID; ++kk) {
    float zv = zr[kk];
#pragma unroll
    for (int e = 0; e < NEXP; ++e) lg[e] += zv * Wr[kk * NEXP + e];
  }
  float gv = grad[n];
#pragma unroll
  for (int e = 0; e < NEXP; ++e) lg[e] += gv * Wr[HID * NEXP + e];
  float mx = fmaxf(fmaxf(lg[0], lg[1]), fmaxf(lg[2], lg[3]));
  float p[NEXP]; float sum = 0.f;
#pragma unroll
  for (int e = 0; e < NEXP; ++e) { p[e] = expf(lg[e] - mx); sum += p[e]; }
  int best = 0; float bp = -1.f;
#pragma unroll
  for (int e = 0; e < NEXP; ++e) {
    p[e] /= sum;
    probs_out[(size_t)n * NEXP + e] = p[e];
    if (p[e] > bp) { bp = p[e]; best = e; }
  }
#pragma unroll
  for (int e = 0; e < NEXP; ++e) gate[(size_t)e * NTOK + n] = (e == best) ? bp : 0.f;
}

__global__ void repack_wd2_kernel(const float* __restrict__ Wd2, const float* __restrict__ bd2,
                                  float* __restrict__ We, float* __restrict__ be)
{
  int i = blockIdx.x * blockDim.x + threadIdx.x;
  if (i < HID * NGENE) {
    int kk = i / NGENE, gg = i % NGENE;
    We[i] = Wd2[(size_t)kk * (2 * NGENE) + 2 * gg];   // even (mu) columns only
  }
  if (i < NGENE) be[i] = bd2[2 * i];
}

__global__ void funchead_kernel(const float* __restrict__ t, const float* __restrict__ Wf2,
                                const float* __restrict__ bf2, float* __restrict__ gout)
{
  int n = blockIdx.x * blockDim.x + threadIdx.x;
  if (n >= NTOK) return;
  float s = bf2[0];
  const float* tr = t + (size_t)n * 64;
#pragma unroll 8
  for (int kk = 0; kk < 64; ++kk) s += tr[kk] * Wf2[kk];
  gout[n] = 1.0f / (1.0f + expf(-s));
}

// ---------------------------------------------------------------------------
static inline void gemm(hipStream_t s, const float* A, const float* W, const float* bias,
                        const float* resid, const float* rowscale, float* C,
                        int M, int N, int K, int act, int accum)
{
  dim3 g(M / 64, (N + 31) / 32);
  gemm_bf16_kernel<<<g, 128, 0, s>>>(A, W, bias, resid, rowscale, C, M, N, K, act, accum);
}

extern "C" void kernel_launch(void* const* d_in, const int* in_sizes, int n_in,
                              void* d_out, int out_size, void* d_ws, size_t ws_size,
                              hipStream_t stream)
{
  (void)in_sizes; (void)n_in; (void)out_size; (void)ws_size;
  const float* vis   = (const float*)d_in[0];
  const float* pos   = (const float*)d_in[1];
  const float* grad  = (const float*)d_in[2];
  const float* Bfour = (const float*)d_in[3];
  const float* W_img = (const float*)d_in[4];
  const float* b_img = (const float*)d_in[5];
  const float* W_pos = (const float*)d_in[6];
  const float* b_pos = (const float*)d_in[7];
  const float* Wq = (const float*)d_in[8];   const float* bq = (const float*)d_in[9];
  const float* Wk = (const float*)d_in[10];  const float* bk = (const float*)d_in[11];
  const float* Wv = (const float*)d_in[12];  const float* bv = (const float*)d_in[13];
  const float* Wo = (const float*)d_in[14];  const float* bo = (const float*)d_in[15];
  const float* ln1_g = (const float*)d_in[16]; const float* ln1_b = (const float*)d_in[17];
  const float* Wr = (const float*)d_in[18];  const float* br = (const float*)d_in[19];
  const float* We1 = (const float*)d_in[20]; const float* be1 = (const float*)d_in[21];
  const float* We2 = (const float*)d_in[22]; const float* be2 = (const float*)d_in[23];
  const float* Wd1 = (const float*)d_in[24]; const float* bd1 = (const float*)d_in[25];
  const float* lnd_g = (const float*)d_in[26]; const float* lnd_b = (const float*)d_in[27];
  const float* Wd2 = (const float*)d_in[28]; const float* bd2 = (const float*)d_in[29];
  const float* Wf1 = (const float*)d_in[30]; const float* bf1 = (const float*)d_in[31];
  const float* Wf2 = (const float*)d_in[32]; const float* bf2 = (const float*)d_in[33];

  float* ws = (float*)d_ws;
  size_t off = 0;
  auto alloc = [&](size_t n) { float* p = ws + off; off += n; return p; };
  float* four = alloc((size_t)NTOK * 128);
  float* z    = alloc((size_t)NTOK * HID);
  float* q    = alloc((size_t)NTOK * HID);
  float* k    = alloc((size_t)NTOK * HID);
  float* v    = alloc((size_t)NTOK * HID);
  float* ao   = alloc((size_t)NTOK * HID);
  float* z2   = alloc((size_t)NTOK * HID);
  float* zmoe = alloc((size_t)NTOK * HID);
  float* hbuf = alloc((size_t)NTOK * 1024);
  float* gate = alloc((size_t)NEXP * NTOK);
  float* dtmp = alloc((size_t)NTOK * HID);
  float* t64  = alloc((size_t)NTOK * 64);
  float* wd2e = alloc((size_t)HID * NGENE);
  float* bd2e = alloc((size_t)NGENE);

  float* mu_out    = (float*)d_out;
  float* g_out     = mu_out + (size_t)NTOK * NGENE;
  float* probs_out = g_out + NTOK;

  // 1) positional Fourier features + fused input projection
  fourier_kernel<<<(NTOK * 64 + 255) / 256, 256, 0, stream>>>(pos, Bfour, four);
  gemm(stream, vis,  W_img, b_img, nullptr, nullptr, z, NTOK, HID, DUIN, 0, 0);
  gemm(stream, four, W_pos, b_pos, nullptr, nullptr, z, NTOK, HID, 128,  0, 1);

  // 2) attention
  gemm(stream, z, Wq, bq, nullptr, nullptr, q, NTOK, HID, HID, 0, 0);
  gemm(stream, z, Wk, bk, nullptr, nullptr, k, NTOK, HID, HID, 0, 0);
  gemm(stream, z, Wv, bv, nullptr, nullptr, v, NTOK, HID, HID, 0, 0);
  attn_kernel<<<dim3(NTOK / 64, NHEAD), 128, 0, stream>>>(q, k, v, ao);
  gemm(stream, ao, Wo, bo, z, nullptr, z2, NTOK, HID, HID, 0, 0);   // +residual z
  ln_kernel<<<NTOK / 8, 256, 0, stream>>>(z2, ln1_g, ln1_b, NTOK, 0);

  // 3) MoE (top-1): router then gated expert accumulation into zmoe
  router_kernel<<<(NTOK + 255) / 256, 256, 0, stream>>>(z2, grad, Wr, br, probs_out, gate);
  hipMemcpyAsync(zmoe, z2, (size_t)NTOK * HID * sizeof(float),
                 hipMemcpyDeviceToDevice, stream);
  for (int e = 0; e < NEXP; ++e) {
    gemm(stream, z2, We1 + (size_t)e * HID * 1024, be1 + (size_t)e * 1024,
         nullptr, nullptr, hbuf, NTOK, 1024, HID, 1, 0);            // GELU
    gemm(stream, hbuf, We2 + (size_t)e * 1024 * HID, be2 + (size_t)e * HID,
         nullptr, gate + (size_t)e * NTOK, zmoe, NTOK, HID, 1024, 0, 1); // gated +=
  }

  // 4) gene decoder -> mu (even columns of Wd2 only)
  gemm(stream, zmoe, Wd1, bd1, nullptr, nullptr, dtmp, NTOK, HID, HID, 0, 0);
  ln_kernel<<<NTOK / 8, 256, 0, stream>>>(dtmp, lnd_g, lnd_b, NTOK, 1); // LN + GELU
  repack_wd2_kernel<<<(HID * NGENE + 255) / 256, 256, 0, stream>>>(Wd2, bd2, wd2e, bd2e);
  gemm(stream, dtmp, wd2e, bd2e, nullptr, nullptr, mu_out, NTOK, NGENE, HID, 2, 0); // softplus

  // 5) func head
  gemm(stream, zmoe, Wf1, bf1, nullptr, nullptr, t64, NTOK, 64, HID, 1, 0);         // GELU
  funchead_kernel<<<(NTOK + 255) / 256, 256, 0, stream>>>(t64, Wf2, bf2, g_out);
}
